// Quantize_34153579937987
// MI455X (gfx1250) — matrix operand order repaired and checked by reading, not compiled, not run on hardware
//
#include <hip/hip_runtime.h>
#include <hip/hip_bf16.h>
#include <math.h>

// ---------------------------------------------------------------------------
// Problem dims (from reference): B=32768, C=256, N_EMBED=8192
// ---------------------------------------------------------------------------
#define BDIM 32768
#define CDIM 256
#define NDIM 8192
#define DECAY 0.99f
#define ONE_MINUS_DECAY 0.01f
#define EPSV 1e-5f

// WMMA tiling
#define BM 64            // rows of inputs per workgroup (A tile -> registers)
#define BN 64            // embed columns per chunk (B tile, TDM double-buffered)
#define NCHUNK (NDIM / BN)

typedef __attribute__((ext_vector_type(16))) __bf16 v16bf;
typedef __attribute__((ext_vector_type(8)))  float  v8f;
typedef __attribute__((ext_vector_type(4)))  unsigned int u32x4;
typedef __attribute__((ext_vector_type(8)))  int    i32x8;
typedef __attribute__((ext_vector_type(4)))  int    i32x4;

#ifndef __has_builtin
#define __has_builtin(x) 0
#endif
#if __has_builtin(__builtin_amdgcn_tensor_load_to_lds) && __has_builtin(__builtin_amdgcn_s_wait_tensorcnt)
#define HAVE_TDM 1
#else
#define HAVE_TDM 0
#endif
#if __has_builtin(__builtin_amdgcn_sched_group_barrier)
#define HAVE_SCHED 1
#else
#define HAVE_SCHED 0
#endif

static __device__ __forceinline__ unsigned short f2bf(float f) {
  union { float f; unsigned u; } v; v.f = f;
  unsigned u = v.u;
  // round-to-nearest-even f32 -> bf16
  unsigned r = (u + 0x7FFFu + ((u >> 16) & 1u)) >> 16;
  return (unsigned short)r;
}

static __device__ __forceinline__ void waveReduceAtomicAdd(float v, float* dst) {
  #pragma unroll
  for (int off = 16; off > 0; off >>= 1) v += __shfl_xor(v, off, 32);
  if ((threadIdx.x & 31) == 0) atomicAdd(dst, v);
}

#if HAVE_TDM
// Issue one TDM transfer: [BN rows][CDIM cols] of bf16, row stride CDIM,
// global -> LDS offset ldsOff. Descriptor per CDNA5 ISA sec 8.
static __device__ __forceinline__ void tdm_issue(unsigned ldsOff, const void* gptr) {
  unsigned long long ga = (unsigned long long)gptr;
  u32x4 g0;
  g0[0] = 1u;                                       // count=1, user mode
  g0[1] = ldsOff;                                   // lds_addr
  g0[2] = (unsigned)(ga & 0xFFFFFFFFull);           // global_addr[31:0]
  g0[3] = (unsigned)((ga >> 32) & 0x01FFFFFFull) | (2u << 30); // addr[56:32]|type=2
  i32x8 g1;
  g1[0] = (int)(1u << 16);                          // data_size=1 (2 bytes)
  g1[1] = (int)((unsigned)CDIM << 16);              // tensor_dim0[15:0] @ bit48
  g1[2] = (int)((unsigned)NDIM << 16);              // tensor_dim1[15:0] @ bit80
  g1[3] = (int)((unsigned)CDIM << 16);              // tile_dim0 @ bit112
  g1[4] = BN;                                       // tile_dim1 (tile_dim2=0)
  g1[5] = CDIM;                                     // tensor_dim0_stride[31:0]
  g1[6] = 0;
  g1[7] = 0;
  i32x4 gz;  gz[0] = 0; gz[1] = 0; gz[2] = 0; gz[3] = 0;
  i32x8 gz8; gz8[0] = 0; gz8[1] = 0; gz8[2] = 0; gz8[3] = 0;
  gz8[4] = 0; gz8[5] = 0; gz8[6] = 0; gz8[7] = 0;
  __builtin_amdgcn_tensor_load_to_lds(g0, g1, gz, gz, gz8, 0);
}
#endif

// ---------------------------------------------------------------------------
// Kernel 1: transpose embed [C][N] -> [N][C] (f32 + bf16), and ||e_n||^2
// ---------------------------------------------------------------------------
__global__ __launch_bounds__(256) void prep_kernel(
    const float* __restrict__ embed,          // [C][N]
    float* __restrict__ embT,                 // [N][C]
    unsigned short* __restrict__ embTb,       // [N][C] bf16
    float* __restrict__ norme)                // [N]
{
  int n = blockIdx.x * 256 + threadIdx.x;
  float s = 0.f;
  for (int c = 0; c < CDIM; ++c) {
    float v = embed[(size_t)c * NDIM + n];    // coalesced across threads
    s += v * v;
    embT [(size_t)n * CDIM + c] = v;
    embTb[(size_t)n * CDIM + c] = f2bf(v);
  }
  norme[n] = s;
}

// ---------------------------------------------------------------------------
// Kernel 2: WMMA score GEMM + row argmin of (||e||^2 - 2 x.e)
//   grid = B/BM workgroups, 256 threads (8 wave32 waves).
//   A tile (64x256 bf16) staged once through LDS into registers.
//   B tiles double-buffered in LDS via the Tensor Data Mover, overlapped
//   with compute (s_wait_tensorcnt(1)).  sched_group_barrier pins a
//   DS-read/WMMA interleave with one k-step of load lookahead.
// ---------------------------------------------------------------------------
__global__ __launch_bounds__(256) void argmin_gemm_kernel(
    const float* __restrict__ inputs,         // [B][C]
    const unsigned short* __restrict__ embbf, // [N][C] bf16
    const float* __restrict__ norme,          // [N]
    int* __restrict__ indOut)                 // [B]
{
  __shared__ __align__(16) unsigned char smem[2 * BN * CDIM * 2]; // 64KB
  unsigned short* buf0 = (unsigned short*)smem;                   // LDS off 0
  unsigned short* buf1 = (unsigned short*)(smem + BN * CDIM * 2); // LDS off 32768

  const int tid    = threadIdx.x;
  const int lane   = tid & 31;
  const int w      = tid >> 5;        // wave id 0..7
  const int lanelo = lane & 15;
  const int lanehi = lane >> 4;       // 0 or 1
  const int rowBase = blockIdx.x * BM;

  // wave -> (m block, n half): waves 0..3 m=0..3 (n half 0), waves 4..7 n half 1
  const int mb     = (w & 3) * 16;
  const int nbHalf = (w >> 2) * 32;

  union Frag { v16bf v; uint4 u[2]; };

  // ---- stage A tile through buf0, hoist this wave's 8 fragments to VGPRs ----
  Frag aF[8];
  {
    unsigned short* sA = buf0;        // 64x256 bf16 = 32KB
    const float* gA = inputs + (size_t)rowBase * CDIM;
    for (int i = tid; i < BM * CDIM; i += 256) sA[i] = f2bf(gA[i]);
    __syncthreads();
    #pragma unroll
    for (int kk = 0; kk < 8; ++kk) {
      const int k0   = kk * 32;
      const int arow = mb + lanelo;
      const int acb  = k0 + (lanehi << 3);
      aF[kk].u[0] = *(const uint4*)&sA[arow * CDIM + acb];
      aF[kk].u[1] = *(const uint4*)&sA[arow * CDIM + acb + 16];
    }
    __syncthreads();                  // A staging done; buf0 free for TDM
  }

  float best[8];
  int   bidx[8];
  #pragma unroll
  for (int v = 0; v < 8; ++v) { best[v] = 3.4e38f; bidx[v] = 0; }

  const int bn0 = nbHalf + lanelo;    // B column (chunk-local) for acc0
  const int bn1 = bn0 + 16;           // B column for acc1
  const int kbo = lanehi << 3;        // per-lane K sub-offset

#if HAVE_TDM
  if (tid == 0) tdm_issue(0u, embbf);            // chunk 0 -> buf0
#endif

  for (int chunk = 0; chunk < NCHUNK; ++chunk) {
    const int cur = chunk & 1;
    const unsigned short* sB = cur ? buf1 : buf0;

#if HAVE_TDM
    // prefetch next chunk into the other buffer (free since chunk-1's barrier)
    if (chunk + 1 < NCHUNK) {
      if (tid == 0)
        tdm_issue(cur ? 0u : (unsigned)(BN * CDIM * 2),
                  embbf + (size_t)(chunk + 1) * BN * CDIM);
      __builtin_amdgcn_s_wait_tensorcnt(1);      // current chunk landed
    } else {
      __builtin_amdgcn_s_wait_tensorcnt(0);      // drain last transfer
    }
    __syncthreads();
#else
    {
      unsigned short* dst = cur ? buf1 : buf0;
      const uint4* src4 = (const uint4*)(embbf + (size_t)chunk * BN * CDIM);
      uint4* dst4 = (uint4*)dst;
      #pragma unroll
      for (int i = 0; i < (BN * CDIM) / (8 * 256); ++i)
        dst4[tid + i * 256] = src4[tid + i * 256];
      __syncthreads();
    }
#endif

    v8f acc0, acc1;
    #pragma unroll
    for (int v = 0; v < 8; ++v) { acc0[v] = 0.f; acc1[v] = 0.f; }

    // K loop over C=256 in steps of 32 (fully unrolled: 32 ds_loads, 16 wmma)
    #pragma unroll
    for (int kk = 0; kk < 8; ++kk) {
      const int kb = kk * 32 + kbo;
      Frag b0, b1;
      b0.u[0] = *(const uint4*)&sB[bn0 * CDIM + kb];
      b0.u[1] = *(const uint4*)&sB[bn0 * CDIM + kb + 16];
      b1.u[0] = *(const uint4*)&sB[bn1 * CDIM + kb];
      b1.u[1] = *(const uint4*)&sB[bn1 * CDIM + kb + 16];
      acc0 = __builtin_amdgcn_wmma_f32_16x16x32_bf16(
          false, aF[kk].v, false, b0.v, (short)0, acc0, false, false);
      acc1 = __builtin_amdgcn_wmma_f32_16x16x32_bf16(
          false, aF[kk].v, false, b1.v, (short)0, acc1, false, false);
    }

#if HAVE_SCHED
    // Pin issue order: one k-step of DS-read lookahead ahead of the WMMAs.
    //   DS x8 (k0,k1) ; 6 x [ WMMA x2 , DS x4 ] ; WMMA x4
    // mask: 0x100 = DS read, 0x008 = MFMA/WMMA
    __builtin_amdgcn_sched_group_barrier(0x100, 8, 0);
    #pragma unroll
    for (int g = 0; g < 6; ++g) {
      __builtin_amdgcn_sched_group_barrier(0x008, 2, 0);
      __builtin_amdgcn_sched_group_barrier(0x100, 4, 0);
    }
    __builtin_amdgcn_sched_group_barrier(0x008, 4, 0);
#endif

    // fold into running argmin:  dist_eff = ||e_n||^2 - 2 * (x . e_n)
    {
      const int n0g = chunk * BN + nbHalf + lanelo;
      const int n1g = n0g + 16;
      const float ne0 = norme[n0g];
      const float ne1 = norme[n1g];
      #pragma unroll
      for (int v = 0; v < 8; ++v) {
        float d0 = ne0 - 2.0f * acc0[v];
        if (d0 < best[v]) { best[v] = d0; bidx[v] = n0g; }
        float d1 = ne1 - 2.0f * acc1[v];
        if (d1 < best[v]) { best[v] = d1; bidx[v] = n1g; }
      }
    }
    __syncthreads();   // all waves done reading sB before it is re-targeted
  }

  // ---- cross-lane argmin (16 lanes share a row), tie-break to lower index ----
  #pragma unroll
  for (int off = 1; off < 16; off <<= 1) {
    #pragma unroll
    for (int v = 0; v < 8; ++v) {
      float ov = __shfl_xor(best[v], off, 32);
      int   oi = __shfl_xor(bidx[v], off, 32);
      if (ov < best[v] || (ov == best[v] && oi < bidx[v])) { best[v] = ov; bidx[v] = oi; }
    }
  }

  // ---- cross-wave merge (wave w and w+4 share rows), reuse LDS ----
  float* sBest = (float*)smem;                 // [2][BM]
  int*   sIdx  = (int*)(smem + 2 * BM * 4);    // [2][BM]
  if (lanelo == 0) {
    const int half = w >> 2;
    #pragma unroll
    for (int v = 0; v < 8; ++v) {
      const int r = mb + 8 * lanehi + v;
      sBest[half * BM + r] = best[v];
      sIdx [half * BM + r] = bidx[v];
    }
  }
  __syncthreads();
  if (tid < BM) {
    float v0 = sBest[tid], v1 = sBest[BM + tid];
    int   i0 = sIdx[tid],  i1 = sIdx[BM + tid];
    int fin = (v1 < v0 || (v1 == v0 && i1 < i0)) ? i1 : i0;
    indOut[rowBase + tid] = fin;
  }
}

// ---------------------------------------------------------------------------
// Kernel 3: per-row scatter (histogram + embed_sum in [N][C]) + quantize/diff
// ---------------------------------------------------------------------------
__global__ __launch_bounds__(256) void scatter_quantize_kernel(
    const float* __restrict__ inputs,   // [B][C]
    const float* __restrict__ embT,     // [N][C]
    const int*   __restrict__ ind,      // [B]
    float* __restrict__ cnt,            // [N]
    float* __restrict__ esumT,          // [N][C]
    float* __restrict__ outQ,           // [B][C]
    float* __restrict__ outDiff,        // [B][C]
    float* __restrict__ outInd)         // [B]
{
  const int b = blockIdx.x;
  const int c = threadIdx.x;
  const int k = ind[b];
  const float x = inputs[(size_t)b * CDIM + c];
  atomicAdd(&esumT[(size_t)k * CDIM + c], x);          // coalesced f32 atomics
  const float q = embT[(size_t)k * CDIM + c];
  outQ  [(size_t)b * CDIM + c] = x + (q - x);          // straight-through
  const float d = q - x;
  outDiff[(size_t)b * CDIM + c] = d * d;
  if (c == 0) { atomicAdd(&cnt[k], 1.0f); outInd[b] = (float)k; }
}

// ---------------------------------------------------------------------------
// Kernel 4: cluster-size EMA + scalar partials (sum, entropy, used, usage)
// ---------------------------------------------------------------------------
__global__ __launch_bounds__(256) void ema1_kernel(
    const float* __restrict__ cs,       // [N] cluster_size (old)
    const float* __restrict__ cnt,      // [N] onehot_sum
    float* __restrict__ outNCS,         // [N] new_cluster_size
    float* __restrict__ acc)            // [0]=sum_ncs [1]=ent [2]=used [3]=usage
{
  const int n = blockIdx.x * 256 + threadIdx.x;
  const float oc  = cnt[n];
  const float ncs = cs[n] * DECAY + ONE_MINUS_DECAY * oc;
  outNCS[n] = ncs;
  const float p = oc * (1.0f / (float)BDIM);           // sum(onehot_sum) == B
  waveReduceAtomicAdd(ncs, &acc[0]);
  waveReduceAtomicAdd(p * __logf(p + 1e-8f), &acc[1]);
  waveReduceAtomicAdd((oc  >= 1.0f) ? 1.f : 0.f, &acc[2]);
  waveReduceAtomicAdd((ncs >= 1.0f) ? 1.f : 0.f, &acc[3]);
}

// ---------------------------------------------------------------------------
// Kernel 5: embed_avg EMA, normalized embed, dk partial
// ---------------------------------------------------------------------------
__global__ __launch_bounds__(256) void ema2_kernel(
    const float* __restrict__ embed,     // [C][N]
    const float* __restrict__ embAvg,    // [C][N]
    const float* __restrict__ esumT,     // [N][C]
    const float* __restrict__ ncs,       // [N] new_cluster_size (from d_out)
    const float* __restrict__ acc,       // [0]=n_total
    float* __restrict__ outEmbAvg,       // [C][N]
    float* __restrict__ outEmb,          // [C][N]
    float* __restrict__ accDk)
{
  const unsigned idx = blockIdx.x * 256 + threadIdx.x;   // < C*N
  const int c = idx / NDIM;
  const int n = idx % NDIM;
  const float ea = embAvg[idx] * DECAY +
                   ONE_MINUS_DECAY * esumT[(size_t)n * CDIM + c];
  outEmbAvg[idx] = ea;
  const float nt  = acc[0];
  const float csn = (ncs[n] + EPSV) / (nt + (float)NDIM * EPSV) * nt;
  const float ne  = ea / csn;
  outEmb[idx] = ne;
  const float d = embed[idx] - ne;
  waveReduceAtomicAdd(d * d, accDk);
}

// ---------------------------------------------------------------------------
// Kernel 6: scalar finalize
// ---------------------------------------------------------------------------
__global__ void finalize_kernel(const float* __restrict__ acc,
                                float* __restrict__ oEnt,
                                float* __restrict__ oUsed,
                                float* __restrict__ oUsage,
                                float* __restrict__ oDk)
{
  *oEnt   = -acc[1];
  *oUsed  = acc[2];
  *oUsage = acc[3];
  *oDk    = sqrtf(acc[4]) / sqrtf((float)CDIM * (float)NDIM);
}

// ---------------------------------------------------------------------------
// Launcher
// ---------------------------------------------------------------------------
extern "C" void kernel_launch(void* const* d_in, const int* in_sizes, int n_in,
                              void* d_out, int out_size, void* d_ws, size_t ws_size,
                              hipStream_t stream) {
  (void)in_sizes; (void)n_in; (void)out_size; (void)ws_size;
  const float* inputs  = (const float*)d_in[0];   // [B][C]
  const float* embed   = (const float*)d_in[1];   // [C][N]
  const float* csize   = (const float*)d_in[2];   // [N]
  const float* embAvg  = (const float*)d_in[3];   // [C][N]

  // workspace layout (bytes)
  unsigned char* ws = (unsigned char*)d_ws;
  float*          embT  = (float*)(ws);                      //  8,388,608 B
  unsigned short* embTb = (unsigned short*)(ws + 8388608);   //  4,194,304 B
  float*          norme = (float*)(ws + 12582912);           //     32,768 B
  int*            ind   = (int*)(ws + 12615680);             //    131,072 B
  float*          cnt   = (float*)(ws + 12746752);           //     32,768 B
  float*          esumT = (float*)(ws + 12779520);           //  8,388,608 B
  float*          acc   = (float*)(ws + 21168128);           //  acc[0..4]

  // zero accumulators every launch (cnt + esumT + acc are contiguous)
  (void)hipMemsetAsync(ws + 12746752, 0, 32768 + 8388608 + 64, stream);

  // output layout (flat f32 concat, reference return order)
  float* out    = (float*)d_out;
  float* oQ     = out;                 // [B*C]
  float* oDiff  = out + 8388608;       // [B*C]
  float* oInd   = out + 16777216;      // [B]
  float* oEnt   = out + 16809984;
  float* oUsed  = out + 16809985;
  float* oUsage = out + 16809986;
  float* oDk    = out + 16809987;
  float* oEmb   = out + 16809988;      // [C*N]
  float* oNCS   = out + 18907140;      // [N]
  float* oEA    = out + 18915332;      // [C*N]

  prep_kernel<<<NDIM / 256, 256, 0, stream>>>(embed, embT, embTb, norme);
  argmin_gemm_kernel<<<BDIM / BM, 256, 0, stream>>>(inputs, embTb, norme, ind);
  scatter_quantize_kernel<<<BDIM, 256, 0, stream>>>(inputs, embT, ind, cnt,
                                                    esumT, oQ, oDiff, oInd);
  ema1_kernel<<<NDIM / 256, 256, 0, stream>>>(csize, cnt, oNCS, acc);
  ema2_kernel<<<(CDIM * NDIM) / 256, 256, 0, stream>>>(embed, embAvg, esumT,
                                                       oNCS, acc, oEA, oEmb,
                                                       acc + 4);
  finalize_kernel<<<1, 1, 0, stream>>>(acc, oEnt, oUsed, oUsage, oDk);
}
